// DMHA_71640054497374
// MI455X (gfx1250) — matrix operand order repaired and checked by reading, not compiled, not run on hardware
//
#include <hip/hip_runtime.h>
#include <hip/hip_bf16.h>

typedef __bf16 bf16;
typedef __attribute__((ext_vector_type(16))) __bf16 v16bf;
typedef __attribute__((ext_vector_type(8)))  __bf16 v8bf;
typedef __attribute__((ext_vector_type(8)))  float  v8f;

#define BATCH 256
#define SEQ   289
#define HEADS 8
#define HD    32
#define DIMV  512
#define CDIM  256
#define QKVN  768
#define MROWS (BATCH * SEQ)   // 73984 = 578*128
#define CHUNK 73
#define NCHUNKS 4

union FragAB { v16bf v; v8bf h[2]; };

// A fragment 16x32 bf16: lane holds row (lane&15); K halves at off and 16+off, off = lane<16?0:8
__device__ __forceinline__ v16bf frag_a(const bf16* p, int ld, int row0, int k0, int lane) {
    const int r   = row0 + (lane & 15);
    const int off = (lane < 16) ? 0 : 8;
    FragAB f;
    f.h[0] = *(const v8bf*)(p + r * ld + k0 + off);
    f.h[1] = *(const v8bf*)(p + r * ld + k0 + 16 + off);
    return f.v;
}

// B fragment 32x16 bf16 from LDS stored [n][k]: lane holds column (lane&15);
// K halves at off and 16+off, off = lane<16?0:8
__device__ __forceinline__ v16bf frag_b(const bf16* p, int ld, int n0, int k0, int lane) {
    const int c   = n0 + (lane & 15);
    const int off = (lane < 16) ? 0 : 8;
    FragAB f;
    f.h[0] = *(const v8bf*)(p + c * ld + k0 + off);
    f.h[1] = *(const v8bf*)(p + c * ld + k0 + 16 + off);
    return f.v;
}

#define WMMA_BF16(A, B, C) \
    __builtin_amdgcn_wmma_f32_16x16x32_bf16(false, (A), false, (B), (short)0, (C), false, false)

// ---------------------------------------------------------------------------
// Kernel 1: W_fusedT[n][k] = (W_reduce (512x256) @ W_qkv (256x768))^T -> bf16 (768x512)
// stored TRANSPOSED so GEMM B-staging is fully vectorized
// ---------------------------------------------------------------------------
__global__ __launch_bounds__(256) void fuse_weights(const float* __restrict__ Wr,
                                                    const float* __restrict__ Wq,
                                                    bf16* __restrict__ WfT) {
    int idx = blockIdx.x * 256 + threadIdx.x;
    if (idx >= DIMV * QKVN) return;
    int i = idx / QKVN, j = idx - i * QKVN;   // i = k-dim (512), j = n-dim (768)
    float acc = 0.f;
    for (int k = 0; k < CDIM; ++k)
        acc += Wr[i * CDIM + k] * Wq[k * QKVN + j];
    WfT[(size_t)j * DIMV + i] = (bf16)acc;
}

// W_projT[n][k] bf16 (512x256) from W_proj (256x512) f32
__global__ __launch_bounds__(256) void cvt_transpose(const float* __restrict__ s,
                                                     bf16* __restrict__ dT) {
    int idx = blockIdx.x * 256 + threadIdx.x;
    if (idx >= CDIM * DIMV) return;
    int k = idx / DIMV, n = idx - k * DIMV;
    dT[(size_t)n * CDIM + k] = (bf16)s[idx];
}

// ---------------------------------------------------------------------------
// Kernel 2: QKV = X (73984x512 f32) @ W_fused (512x768, stored T as 768x512 bf16)
// BM=128 BN=128 BK=32; 256 threads = 8 waves (4x2); wave tile 32x64 (2x4 frags)
// epilogue: branch scaling, scatter bf16 into qkv[3][B][H][SEQ][32]
// ---------------------------------------------------------------------------
__global__ __launch_bounds__(256) void gemm1_qkv(const float* __restrict__ X,
                                                 const bf16* __restrict__ WfT,
                                                 bf16* __restrict__ qkv,
                                                 const float* __restrict__ wscale,
                                                 const float* __restrict__ wbias,
                                                 int mode) {
    __shared__ __align__(16) bf16 As[128 * 32];   // [m][k]
    __shared__ __align__(16) bf16 Bs[128 * 32];   // [n][k]

    const int tid  = threadIdx.x;
    const int lane = tid & 31;
    const int wave = tid >> 5;
    const int bm = blockIdx.x * 128;
    const int bn = blockIdx.y * 128;
    const int wm = (wave >> 1) * 32;
    const int wn = (wave & 1) * 64;

    v8f acc[2][4] = {};

    const int am  = tid >> 1;            // A stage: row, 2 threads per row
    const int akq = (tid & 1) * 16;      // each thread: 16 floats (4 float4)
    const int brow = tid >> 1;           // B stage: n row of WfT
    const int bko  = (tid & 1) * 16;     // 16 contiguous k

    for (int k0 = 0; k0 < DIMV; k0 += 32) {
        // ---- batched global loads (all in flight before any wait) ----
        float4 fa[4];
        #pragma unroll
        for (int j = 0; j < 4; ++j)
            fa[j] = *(const float4*)(X + (size_t)(bm + am) * DIMV + k0 + akq + j * 4);
        v8bf wb[2];
        #pragma unroll
        for (int j = 0; j < 2; ++j)
            wb[j] = *(const v8bf*)(WfT + (size_t)(bn + brow) * DIMV + k0 + bko + j * 8);
        if (k0 + 32 < DIMV)
            __builtin_prefetch(X + (size_t)(bm + am) * DIMV + k0 + 32 + akq, 0, 3);
        // ---- convert + store to LDS (all vectorized) ----
        #pragma unroll
        for (int j = 0; j < 4; ++j) {
            bf16* dst = &As[am * 32 + akq + j * 4];
            dst[0] = (bf16)fa[j].x; dst[1] = (bf16)fa[j].y;
            dst[2] = (bf16)fa[j].z; dst[3] = (bf16)fa[j].w;
        }
        #pragma unroll
        for (int j = 0; j < 2; ++j)
            *(v8bf*)&Bs[brow * 32 + bko + j * 8] = wb[j];
        __syncthreads();

        v16bf a0 = frag_a(As, 32, wm, 0, lane);
        v16bf a1 = frag_a(As, 32, wm + 16, 0, lane);
        #pragma unroll
        for (int ni = 0; ni < 4; ++ni) {
            v16bf b = frag_b(Bs, 32, wn + ni * 16, 0, lane);
            acc[0][ni] = WMMA_BF16(a0, b, acc[0][ni]);
            acc[1][ni] = WMMA_BF16(a1, b, acc[1][ni]);
        }
        __syncthreads();
    }

    // epilogue: C/D layout -> lane holds col (lane&15), rows r + (lane>=16 ? 8 : 0)
    const int half8 = (lane >= 16) ? 8 : 0;
    const int colL  = lane & 15;
    #pragma unroll
    for (int mi = 0; mi < 2; ++mi)
    #pragma unroll
    for (int ni = 0; ni < 4; ++ni)
    #pragma unroll
    for (int r = 0; r < 8; ++r) {
        int row = bm + wm + mi * 16 + r + half8;
        int col = bn + wn + ni * 16 + colL;
        float v = acc[mi][ni][r];
        int t   = col >> 8;      // 0:q 1:k 2:v
        int rem = col & 255;     // h*32+d
        if (mode == 0) { if (t == 0) v *= wscale[rem]; }
        else           { if (t < 2)  v = v * wscale[rem] + wbias[rem]; }
        int b = row / SEQ;
        int n = row - b * SEQ;
        size_t idx = (((size_t)t * BATCH + b) * HEADS + (rem >> 5)) * (SEQ * HD)
                   + (size_t)n * HD + (rem & 31);
        qkv[idx] = (bf16)v;
    }
}

// ---------------------------------------------------------------------------
// Kernel 3: chunked attention, one wave per (b, h, chunk). M pad 80, K pad 96.
// ---------------------------------------------------------------------------
__global__ __launch_bounds__(32) void attn_chunk(const bf16* __restrict__ qkv,
                                                 bf16* __restrict__ O) {
    __shared__ __align__(16) bf16 Qs[80 * 32];
    __shared__ __align__(16) bf16 Ks[96 * 32];
    __shared__ __align__(16) bf16 Vt[32 * 96];   // transposed [d][kv]
    __shared__ __align__(16) bf16 Ps[80 * 96];   // softmax probs (unnormalized)
    __shared__ float rsum[80];

    const int blk = blockIdx.x;
    const int c = blk & 3;
    const int h = (blk >> 2) & 7;
    const int b = blk >> 5;
    const int s  = c * CHUNK;
    const int nc = min(s + CHUNK, SEQ) - s;
    const int lane = threadIdx.x;

    const size_t plane = ((size_t)b * HEADS + h) * (SEQ * HD);
    const size_t strideT = (size_t)BATCH * HEADS * SEQ * HD;
    const bf16* Qg = qkv + plane;
    const bf16* Kg = qkv + strideT + plane;
    const bf16* Vg = qkv + 2 * strideT + plane;

    for (int r = lane; r < 80; r += 32) {
        if (r < nc) {
            #pragma unroll
            for (int q8 = 0; q8 < 4; ++q8)
                *(v8bf*)&Qs[r * 32 + q8 * 8] = *(const v8bf*)(Qg + (size_t)(s + r) * 32 + q8 * 8);
        } else {
            for (int i = 0; i < 32; ++i) Qs[r * 32 + i] = (bf16)0.0f;
        }
    }
    for (int r = lane; r < 96; r += 32) {
        if (r < nc) {
            bf16 vrow[32];
            #pragma unroll
            for (int q8 = 0; q8 < 4; ++q8) {
                *(v8bf*)&Ks[r * 32 + q8 * 8]  = *(const v8bf*)(Kg + (size_t)(s + r) * 32 + q8 * 8);
                *(v8bf*)&vrow[q8 * 8]         = *(const v8bf*)(Vg + (size_t)(s + r) * 32 + q8 * 8);
            }
            #pragma unroll
            for (int dd = 0; dd < 32; ++dd) Vt[dd * 96 + r] = vrow[dd];
        } else {
            for (int i = 0; i < 32; ++i) { Ks[r * 32 + i] = (bf16)0.0f; Vt[i * 96 + r] = (bf16)0.0f; }
        }
    }
    __syncthreads();

    const float scale = 0.17677669529663687f;   // 32^-0.5
    const int colL  = lane & 15;
    const int half8 = (lane >= 16) ? 8 : 0;

    // Phase 1: S = Q K^T strip by strip, masked softmax, P -> LDS (bf16)
    for (int mi = 0; mi < 5; ++mi) {
        v16bf aq = frag_a(Qs, 32, mi * 16, 0, lane);
        float pv[6][8];
        #pragma unroll
        for (int ni = 0; ni < 6; ++ni) {
            v16bf bk = frag_b(Ks, 32, ni * 16, 0, lane);
            v8f z = {};
            v8f sacc = WMMA_BF16(aq, bk, z);
            #pragma unroll
            for (int r = 0; r < 8; ++r) pv[ni][r] = sacc[r] * scale;
        }
        #pragma unroll
        for (int r = 0; r < 8; ++r) {
            float mx = -1e30f;
            #pragma unroll
            for (int ni = 0; ni < 6; ++ni) {
                int col = ni * 16 + colL;
                float sv = (col < nc) ? pv[ni][r] : -1e30f;
                pv[ni][r] = sv;
                mx = fmaxf(mx, sv);
            }
            #pragma unroll
            for (int off = 8; off >= 1; off >>= 1)      // stays inside 16-lane half (wave32)
                mx = fmaxf(mx, __shfl_xor(mx, off, 32));
            float sum = 0.f;
            #pragma unroll
            for (int ni = 0; ni < 6; ++ni) {
                int col = ni * 16 + colL;
                float p = (col < nc) ? __expf(pv[ni][r] - mx) : 0.f;
                pv[ni][r] = p;
                sum += p;
            }
            #pragma unroll
            for (int off = 8; off >= 1; off >>= 1)
                sum += __shfl_xor(sum, off, 32);
            int row = mi * 16 + r + half8;
            if (colL == 0) rsum[row] = sum;
            #pragma unroll
            for (int ni = 0; ni < 6; ++ni)
                Ps[row * 96 + ni * 16 + colL] = (bf16)pv[ni][r];
        }
    }
    __syncthreads();

    // Phase 2: O = P V   (K padded to 96 = 3 steps of 32)
    for (int mi = 0; mi < 5; ++mi) {
        v8f oacc[2] = {};
        #pragma unroll
        for (int kk = 0; kk < 3; ++kk) {
            v16bf ap = frag_a(Ps, 96, mi * 16, kk * 32, lane);
            #pragma unroll
            for (int ni = 0; ni < 2; ++ni) {
                v16bf bv = frag_b(Vt, 96, ni * 16, kk * 32, lane);
                oacc[ni] = WMMA_BF16(ap, bv, oacc[ni]);
            }
        }
        #pragma unroll
        for (int ni = 0; ni < 2; ++ni)
        #pragma unroll
        for (int r = 0; r < 8; ++r) {
            int m = mi * 16 + r + half8;
            if (m < nc) {
                int dd = ni * 16 + colL;
                float v = oacc[ni][r] / rsum[m];
                O[((size_t)(b * SEQ + s + m)) * CDIM + h * HD + dd] = (bf16)v;
            }
        }
    }
}

// ---------------------------------------------------------------------------
// Kernel 4: Y = O (73984x256 bf16) @ W_proj (256x512, stored T as 512x256 bf16)
// + b_proj -> f32.  BM=128 BN=128 BK=32; 8 waves, wave tile 32x64
// ---------------------------------------------------------------------------
__global__ __launch_bounds__(256) void gemm2_proj(const bf16* __restrict__ Ob,
                                                  const bf16* __restrict__ WpT,
                                                  const float* __restrict__ bias,
                                                  float* __restrict__ Y) {
    __shared__ __align__(16) bf16 As[128 * 32];
    __shared__ __align__(16) bf16 Bs[128 * 32];

    const int tid  = threadIdx.x;
    const int lane = tid & 31;
    const int wave = tid >> 5;
    const int bm = blockIdx.x * 128;
    const int bn = blockIdx.y * 128;
    const int wm = (wave >> 1) * 32;
    const int wn = (wave & 1) * 64;

    v8f acc[2][4] = {};

    const int am  = tid >> 1;            // A stage: 2 threads per row, 16 bf16 each
    const int akq = (tid & 1) * 16;
    const int brow = tid >> 1;           // B stage: n row of WpT
    const int bko  = (tid & 1) * 16;

    for (int k0 = 0; k0 < CDIM; k0 += 32) {
        v8bf fa[2];
        #pragma unroll
        for (int j = 0; j < 2; ++j)
            fa[j] = *(const v8bf*)(Ob + (size_t)(bm + am) * CDIM + k0 + akq + j * 8);
        v8bf wb[2];
        #pragma unroll
        for (int j = 0; j < 2; ++j)
            wb[j] = *(const v8bf*)(WpT + (size_t)(bn + brow) * CDIM + k0 + bko + j * 8);
        if (k0 + 32 < CDIM)
            __builtin_prefetch(Ob + (size_t)(bm + am) * CDIM + k0 + 32 + akq, 0, 3);
        #pragma unroll
        for (int j = 0; j < 2; ++j)
            *(v8bf*)&As[am * 32 + akq + j * 8] = fa[j];
        #pragma unroll
        for (int j = 0; j < 2; ++j)
            *(v8bf*)&Bs[brow * 32 + bko + j * 8] = wb[j];
        __syncthreads();

        v16bf a0 = frag_a(As, 32, wm, 0, lane);
        v16bf a1 = frag_a(As, 32, wm + 16, 0, lane);
        #pragma unroll
        for (int ni = 0; ni < 4; ++ni) {
            v16bf b = frag_b(Bs, 32, wn + ni * 16, 0, lane);
            acc[0][ni] = WMMA_BF16(a0, b, acc[0][ni]);
            acc[1][ni] = WMMA_BF16(a1, b, acc[1][ni]);
        }
        __syncthreads();
    }

    const int half8 = (lane >= 16) ? 8 : 0;
    const int colL  = lane & 15;
    #pragma unroll
    for (int mi = 0; mi < 2; ++mi)
    #pragma unroll
    for (int ni = 0; ni < 4; ++ni)
    #pragma unroll
    for (int r = 0; r < 8; ++r) {
        int row = bm + wm + mi * 16 + r + half8;
        int col = bn + wn + ni * 16 + colL;
        Y[(size_t)row * DIMV + col] = acc[mi][ni][r] + bias[col];
    }
}

// ---------------------------------------------------------------------------
extern "C" void kernel_launch(void* const* d_in, const int* in_sizes, int n_in,
                              void* d_out, int out_size, void* d_ws, size_t ws_size,
                              hipStream_t stream) {
    const float* x1       = (const float*)d_in[0];
    const float* x2       = (const float*)d_in[1];
    const float* W_reduce = (const float*)d_in[2];
    const float* W_qkv    = (const float*)d_in[3];
    const float* W_proj   = (const float*)d_in[4];
    const float* b_proj   = (const float*)d_in[5];
    const float* w_ds     = (const float*)d_in[6];
    const float* w_uds    = (const float*)d_in[7];
    const float* b_uds    = (const float*)d_in[8];
    float* out = (float*)d_out;

    char* ws = (char*)d_ws;
    bf16* WfT = (bf16*)(ws);                        //   786,432 B  (768x512, transposed)
    bf16* WpT = (bf16*)(ws + 786432);               //   262,144 B  (512x256, transposed)
    bf16* qkv = (bf16*)(ws + 1048576);              // 113,639,424 B (3*B*H*SEQ*32)
    bf16* Ob  = (bf16*)(ws + 114688000);            //  37,879,808 B (73984x256)

    fuse_weights<<<(DIMV * QKVN + 255) / 256, 256, 0, stream>>>(W_reduce, W_qkv, WfT);
    cvt_transpose<<<(CDIM * DIMV + 255) / 256, 256, 0, stream>>>(W_proj, WpT);

    dim3 g1(MROWS / 128, QKVN / 128);   // 578 x 6
    dim3 g2(MROWS / 128, DIMV / 128);   // 578 x 4
    const int attn_blocks = BATCH * HEADS * NCHUNKS;   // 8192
    const size_t out_half = (size_t)MROWS * DIMV;      // 37,879,808 floats

    // branch 1 (x1, downscale q by w_ds)
    gemm1_qkv<<<g1, 256, 0, stream>>>(x1, WfT, qkv, w_ds, b_uds, 0);
    attn_chunk<<<attn_blocks, 32, 0, stream>>>(qkv, Ob);
    gemm2_proj<<<g2, 256, 0, stream>>>(Ob, WpT, b_proj, out);

    // branch 2 (x2, affine q,k by w_uds/b_uds) -- reuses qkv/Ob scratch
    gemm1_qkv<<<g1, 256, 0, stream>>>(x2, WfT, qkv, w_uds, b_uds, 1);
    attn_chunk<<<attn_blocks, 32, 0, stream>>>(qkv, Ob);
    gemm2_proj<<<g2, 256, 0, stream>>>(Ob, WpT, b_proj, out + out_half);
}
